// GNNPolicy_18940805775462
// MI455X (gfx1250) — matrix operand order repaired and checked by reading, not compiled, not run on hardware
//
#include <hip/hip_runtime.h>
#include <hip/hip_bf16.h>

#define FEAT 128
#define WAVE 32

typedef __attribute__((ext_vector_type(16))) __bf16 v16bf;
typedef __attribute__((ext_vector_type(8)))  float  v8f;

__device__ inline unsigned pack2bf(float lo, float hi) {
    union { __bf16 b[2]; unsigned u; } p;
    p.b[0] = (__bf16)lo; p.b[1] = (__bf16)hi;
    return p.u;
}

// ---------------------------------------------------------------------------
// Degree / normalization kernels
// ---------------------------------------------------------------------------
__global__ void deg_init(float* __restrict__ deg, int n) {
    int i = blockIdx.x * blockDim.x + threadIdx.x;
    if (i < n) deg[i] = 1.0f;  // self-loop contributes 1 to every node's degree
}

__global__ void deg_edges(const long long* __restrict__ dst, float* __restrict__ deg, int e) {
    int i = blockIdx.x * blockDim.x + threadIdx.x;
    if (i < e) atomicAdd(&deg[(int)dst[i]], 1.0f);
}

__global__ void deg_rsqrt(float* __restrict__ deg, int n) {
    int i = blockIdx.x * blockDim.x + threadIdx.x;
    if (i < n) deg[i] = rsqrtf(deg[i]);  // deg >= 1 always (self loops)
}

// ---------------------------------------------------------------------------
// One-time pack: W (128x128 f32, row-major) -> bf16 B-fragments.
// P[((colTile*4 + kstep)*32 + lane)*16 + i] =
//     bf16(W[kstep*32 + (lane>>4)*16 + i][colTile*16 + (lane&15)])
// 32 KB total, L2/L1 resident for all subsequent tiles.
// ---------------------------------------------------------------------------
__global__ void pack_w_bf16(const float* __restrict__ W, __bf16* __restrict__ P) {
    int t = blockIdx.x * blockDim.x + threadIdx.x;
    if (t >= 8 * 4 * 32) return;
    int lane = t & 31, s = (t >> 5) & 3, colTile = t >> 7;
    int col = colTile * 16 + (lane & 15);
    int k0  = s * 32 + (lane >> 4) * 16;
    __bf16* out = P + (size_t)t * 16;
    #pragma unroll
    for (int i = 0; i < 16; ++i) out[i] = (__bf16)W[(size_t)(k0 + i) * FEAT + col];
}

// ---------------------------------------------------------------------------
// H = X (n x 128) @ W (128 x 128) with bf16 WMMA, f32 accumulate.
// Block = 8 waves = one 16-row stripe; wave w computes colTile w.
// A tile (16x128) staged once per block in LDS, pre-swizzled into the
// per-lane 16-bit A 16x32 fragment layout; B comes pre-packed in bf16.
// ---------------------------------------------------------------------------
__global__ void gemm_wmma_bf16(const float* __restrict__ X,
                               const __bf16* __restrict__ Wpk,
                               float* __restrict__ H, int n) {
    __shared__ __align__(32) unsigned char ldsA[4 * 32 * 32];  // [kstep][lane][32B]

    const int rowTile = blockIdx.x;
    const int t = threadIdx.x;

    // ---- cooperative fill: 256 threads x 16 bytes = 4 KB (full A tile, bf16)
    {
        int s = t >> 6;            // kstep 0..3
        int r = (t >> 5) & 1;      // which 8-value half of the fragment
        int l = t & 31;            // fragment lane
        int half = l >> 4, m = l & 15;
        int row = rowTile * 16 + m;
        if (row >= n) row = n - 1;                       // clamp (harmless dup)
        const float* xp = X + (size_t)row * FEAT + s * 32 + half * 16 + r * 8;
        uint4 v;
        v.x = pack2bf(xp[0], xp[1]);
        v.y = pack2bf(xp[2], xp[3]);
        v.z = pack2bf(xp[4], xp[5]);
        v.w = pack2bf(xp[6], xp[7]);
        *(uint4*)&ldsA[((s * 32 + l) << 5) + (r << 4)] = v;
    }
    __syncthreads();

    const int colTile = t >> 5;        // 8 waves -> 8 column tiles
    const int lane    = t & 31;
    const int half    = lane >> 4;
    const int m       = lane & 15;
    const int col     = colTile * 16 + m;

    const v16bf* Bp = (const v16bf*)Wpk + (size_t)colTile * 4 * 32;

    v8f acc = {};
    #pragma unroll
    for (int s = 0; s < 4; ++s) {
        v16bf a = *(const v16bf*)&ldsA[((s * 32 + lane) << 5)];   // 2x ds_load_b128
        v16bf b = Bp[s * 32 + lane];                               // 32B contiguous
        acc = __builtin_amdgcn_wmma_f32_16x16x32_bf16(
                  false, a, false, b, (short)0, acc, false, false);
    }

    // ---- C store: VGPR r -> row (tile + half*8 + r), column `col`.
    // Block-uniform bounds test -> unpredicated store clause in the common case.
    float* outp = H + (size_t)(rowTile * 16 + half * 8) * FEAT + col;
    if ((rowTile + 1) * 16 <= n) {
        #pragma unroll
        for (int r = 0; r < 8; ++r) outp[(size_t)r * FEAT] = acc[r];
    } else {
        int rowBase = rowTile * 16 + half * 8;
        #pragma unroll
        for (int r = 0; r < 8; ++r)
            if (rowBase + r < n) outp[(size_t)r * FEAT] = acc[r];
    }
}

// ---------------------------------------------------------------------------
// Aggregation: out = sum over edges of dis[s]*dis[d]*h[s] (+ self loop)
// ---------------------------------------------------------------------------
__global__ void self_init(float* __restrict__ out, const float* __restrict__ h,
                          const float* __restrict__ dis, int n) {
    size_t i = (size_t)blockIdx.x * blockDim.x + threadIdx.x;
    if (i >= (size_t)n * FEAT) return;
    float d = dis[i >> 7];
    out[i] = h[i] * d * d;          // self-loop message: norm = dis[i]^2
}

__global__ void edge_agg(const long long* __restrict__ src,
                         const long long* __restrict__ dst,
                         const float* __restrict__ dis,
                         const float* __restrict__ h,
                         float* __restrict__ out, int e) {
    int wid  = (blockIdx.x * blockDim.x + threadIdx.x) >> 5;  // one wave per edge
    int lane = threadIdx.x & 31;
    if (wid >= e) return;
    int s = (int)src[wid];
    int d = (int)dst[wid];
    float nrm = dis[s] * dis[d];
    float4 v = ((const float4*)(h + (size_t)s * FEAT))[lane];   // 4 feats/lane
    float* op = out + (size_t)d * FEAT + lane * 4;
    atomicAdd(op + 0, v.x * nrm);   // L2-resident f32 atomics (51MB < 192MB L2)
    atomicAdd(op + 1, v.y * nrm);
    atomicAdd(op + 2, v.z * nrm);
    atomicAdd(op + 3, v.w * nrm);
}

__global__ void bias_relu(float* __restrict__ h, const float* __restrict__ b, int n) {
    size_t i = (size_t)blockIdx.x * blockDim.x + threadIdx.x;
    if (i >= (size_t)n * FEAT) return;
    float v = h[i] + b[i & (FEAT - 1)];
    h[i] = v > 0.0f ? v : 0.0f;
}

// ---------------------------------------------------------------------------
// out[i] = dot(h[i, :], Wh[:, 0]) + bh   (OUT == 1) — wave per node
// ---------------------------------------------------------------------------
__global__ void final_dot(const float* __restrict__ h, const float* __restrict__ Wh,
                          const float* __restrict__ bh, float* __restrict__ out, int n) {
    int wid  = (blockIdx.x * blockDim.x + threadIdx.x) >> 5;
    int lane = threadIdx.x & 31;
    if (wid >= n) return;
    float4 v = ((const float4*)(h + (size_t)wid * FEAT))[lane];
    float4 w = ((const float4*)Wh)[lane];
    float acc = v.x * w.x + v.y * w.y + v.z * w.z + v.w * w.w;
    #pragma unroll
    for (int off = 16; off > 0; off >>= 1)      // wave32 reduction
        acc += __shfl_xor(acc, off, WAVE);
    if (lane == 0) out[wid] = acc + bh[0];
}

// ---------------------------------------------------------------------------
extern "C" void kernel_launch(void* const* d_in, const int* in_sizes, int n_in,
                              void* d_out, int out_size, void* d_ws, size_t ws_size,
                              hipStream_t stream) {
    const float*     X  = (const float*)d_in[0];
    const long long* ei = (const long long*)d_in[1];   // int64 [2, E]
    const float*     W1 = (const float*)d_in[2];
    const float*     b1 = (const float*)d_in[3];
    const float*     W2 = (const float*)d_in[4];
    const float*     b2 = (const float*)d_in[5];
    const float*     Wh = (const float*)d_in[6];
    const float*     bh = (const float*)d_in[7];
    float* out = (float*)d_out;

    const int N = in_sizes[0] / FEAT;
    const int E = in_sizes[1] / 2;
    const long long* src = ei;
    const long long* dst = ei + E;

    // Workspace: dis[Npad] | bufA[N*128] | bufB[N*128] | pkW1[16K bf16] | pkW2[16K bf16]
    float* dis  = (float*)d_ws;
    float* bufA = dis + (((size_t)N + 255) & ~(size_t)255);
    float* bufB = bufA + (size_t)N * FEAT;
    __bf16* pkW1 = (__bf16*)(bufB + (size_t)N * FEAT);
    __bf16* pkW2 = pkW1 + 8 * 4 * 32 * 16;

    const int TB = 256;
    int nBlocksN     = (N + TB - 1) / TB;
    int nBlocksE     = (E + TB - 1) / TB;
    int nBlocksElem  = (int)(((size_t)N * FEAT + TB - 1) / TB);
    int nBlocksGemm  = (N + 15) / 16;                       // one block per 16-row stripe
    int nBlocksEdgeW = (int)(((size_t)E * WAVE + TB - 1) / TB);
    int nBlocksNodeW = (int)(((size_t)N * WAVE + TB - 1) / TB);

    // --- one-time weight packing (bf16, WMMA B-fragment layout) ---
    pack_w_bf16<<<4, TB, 0, stream>>>(W1, pkW1);
    pack_w_bf16<<<4, TB, 0, stream>>>(W2, pkW2);

    // --- symmetric degree normalization (shared by both layers) ---
    deg_init <<<nBlocksN, TB, 0, stream>>>(dis, N);
    deg_edges<<<nBlocksE, TB, 0, stream>>>(dst, dis, E);
    deg_rsqrt<<<nBlocksN, TB, 0, stream>>>(dis, N);

    // --- layer 1: relu(agg(X @ W1) + b1) -> bufB ---
    gemm_wmma_bf16<<<nBlocksGemm, TB, 0, stream>>>(X, pkW1, bufA, N);
    self_init     <<<nBlocksElem, TB, 0, stream>>>(bufB, bufA, dis, N);
    edge_agg      <<<nBlocksEdgeW, TB, 0, stream>>>(src, dst, dis, bufA, bufB, E);
    bias_relu     <<<nBlocksElem, TB, 0, stream>>>(bufB, b1, N);

    // --- layer 2: relu(agg(bufB @ W2) + b2) -> bufB (bufA as temp) ---
    gemm_wmma_bf16<<<nBlocksGemm, TB, 0, stream>>>(bufB, pkW2, bufA, N);
    self_init     <<<nBlocksElem, TB, 0, stream>>>(bufB, bufA, dis, N);
    edge_agg      <<<nBlocksEdgeW, TB, 0, stream>>>(src, dst, dis, bufA, bufB, E);
    bias_relu     <<<nBlocksElem, TB, 0, stream>>>(bufB, b2, N);

    // --- head: out = bufB @ Wh + bh ---
    final_dot<<<nBlocksNodeW, TB, 0, stream>>>(bufB, Wh, bh, out, N);
}
